// SlidingWindowAttention_65395172049081
// MI455X (gfx1250) — matrix operand order repaired and checked by reading, not compiled, not run on hardware
//
#include <hip/hip_runtime.h>
#include <hip/hip_bf16.h>

#define B_   2
#define S_   4096
#define HID  1024
#define NH_  16
#define HD   64
#define WIN  512

typedef __attribute__((ext_vector_type(16))) __bf16 v16bf;
typedef __attribute__((ext_vector_type(8)))  __bf16 v8bf;
typedef __attribute__((ext_vector_type(8)))  float  v8f;

// ---------- helpers ----------
static __device__ __forceinline__ unsigned short f2bf_bits(float f) {
  unsigned u = __builtin_bit_cast(unsigned, f);
  u += 0x7FFFu + ((u >> 16) & 1u);          // round-to-nearest-even
  return (unsigned short)(u >> 16);
}
static __device__ __forceinline__ __bf16 f2bf(float f) {
  unsigned short h = f2bf_bits(f);
  return __builtin_bit_cast(__bf16, h);
}

// A-fragment (16x32 bf16, M x K): lane L(0-15) row L, K 0..15 contiguous;
// lane L+16 same row, K 16..31.  p = rowStart + 16*(lane>=16).
static __device__ __forceinline__ v16bf load_a16(const __bf16* p) {
  v8bf lo = *(const v8bf*)(p);
  v8bf hi = *(const v8bf*)(p + 8);
  return __builtin_shufflevector(lo, hi, 0,1,2,3,4,5,6,7,8,9,10,11,12,13,14,15);
}
// B-fragment (32x16 bf16, K x N): lane L(0-15) col L, K 0..15 contiguous;
// lane L+16 col L, K 16..31.  p = colStart + 16*(lane>=16).
static __device__ __forceinline__ v16bf load_b16(const __bf16* p) {
  return *(const v16bf*)(p);
}
static __device__ __forceinline__ v8f wmma_bf16(v16bf a, v16bf b, v8f c) {
  return __builtin_amdgcn_wmma_f32_16x16x32_bf16(false, a, false, b, (short)0, c,
                                                 false, false);
}

// ---------- kernel 1: f32 -> bf16 conversion (vectorized x4) ----------
__global__ __launch_bounds__(256) void swa_cvt_bf16(const float* __restrict__ in,
                                                    __bf16* __restrict__ out,
                                                    int n4) {
  int i = blockIdx.x * 256 + threadIdx.x;
  if (i >= n4) return;
  float4 v = ((const float4*)in)[i];
  ushort4 o;
  o.x = f2bf_bits(v.x); o.y = f2bf_bits(v.y);
  o.z = f2bf_bits(v.z); o.w = f2bf_bits(v.w);
  ((ushort4*)out)[i] = o;
}

// ---------- kernel 2: fused QKV projection GEMM ----------
// grid = (256, 8), block = 256 (8 waves).
// One wave -> TWO 16-row tiles x one 16-col tile of Q,K,V (B frags reused 2x),
// software-pipelined one k-step ahead.
__global__ __launch_bounds__(256) void swa_qkv_gemm(
    const __bf16* __restrict__ xb,                       // [8192][1024]
    const __bf16* __restrict__ wq, const __bf16* __restrict__ wk,
    const __bf16* __restrict__ wv,                       // [1024][1024] row-major
    __bf16* __restrict__ Qo,   // [b][h][s][64]  (pre-scaled by 1/8)
    __bf16* __restrict__ Ko,   // [b][h][s][64]
    __bf16* __restrict__ Vt)   // [b][h][64][s]  (transposed)
{
  const int wave = threadIdx.x >> 5;
  const int lane = threadIdx.x & 31;
  const int m0 = blockIdx.x * 32;                        // two row tiles
  const int n0 = (blockIdx.y * 8 + wave) * 16;
  const int c    = lane & 15;
  const int hb   = lane >> 4;
  const int ak8  = hb * 16;
  const int bk16 = hb * 16;

  const __bf16* Ap0 = xb + (size_t)(m0 + c) * HID + ak8;
  const __bf16* Ap1 = Ap0 + (size_t)16 * HID;
  const __bf16* Bq  = wq + (size_t)(n0 + c) * HID + bk16;
  const __bf16* Bk  = wk + (size_t)(n0 + c) * HID + bk16;
  const __bf16* Bv  = wv + (size_t)(n0 + c) * HID + bk16;

  v8f cq0 = {}, cq1 = {}, ck0 = {}, ck1 = {}, cv0 = {}, cv1 = {};

  // pipeline prologue
  v16bf a0 = load_a16(Ap0);
  v16bf a1 = load_a16(Ap1);
  v16bf bq = load_b16(Bq);
  v16bf bk = load_b16(Bk);
  v16bf bv = load_b16(Bv);

#pragma unroll 2
  for (int k = 0; k < HID; k += 32) {
    // prefetch next k-step (last-iter prefetch reads 64B past the row into the
    // adjacent ws buffer: in-bounds, unused)
    v16bf a0n = load_a16(Ap0 + k + 32);
    v16bf a1n = load_a16(Ap1 + k + 32);
    v16bf bqn = load_b16(Bq + k + 32);
    v16bf bkn = load_b16(Bk + k + 32);
    v16bf bvn = load_b16(Bv + k + 32);

    cq0 = wmma_bf16(a0, bq, cq0);
    cq1 = wmma_bf16(a1, bq, cq1);
    ck0 = wmma_bf16(a0, bk, ck0);
    ck1 = wmma_bf16(a1, bk, ck1);
    cv0 = wmma_bf16(a0, bv, cv0);
    cv1 = wmma_bf16(a1, bv, cv1);

    a0 = a0n; a1 = a1n; bq = bqn; bk = bkn; bv = bvn;
  }

  const int h  = n0 >> 6;
  const int dc = (n0 & 63) + c;
#pragma unroll
  for (int mt = 0; mt < 2; ++mt) {
    v8f cq = mt ? cq1 : cq0;
    v8f ck = mt ? ck1 : ck0;
    v8f cv = mt ? cv1 : cv0;
#pragma unroll
    for (int rr = 0; rr < 8; ++rr) {
      int m = m0 + mt * 16 + rr + hb * 8;
      int b = m >> 12;                // / 4096
      int s = m & (S_ - 1);
      size_t bh = (size_t)b * NH_ + h;
      Qo[(bh * S_ + s) * HD + dc] = f2bf(cq[rr] * 0.125f);   // 1/sqrt(64)
      Ko[(bh * S_ + s) * HD + dc] = f2bf(ck[rr]);
      Vt[(bh * HD + dc) * S_ + s] = f2bf(cv[rr]);
    }
  }
}

// ---------- kernel 3: sliding-window flash attention ----------
// grid = (32, 32): x -> 8 query tiles per block (1 per wave), y -> b*16+h.
// K fragments for the next 32-key step are prefetched across the back-edge.
__global__ __launch_bounds__(256) void swa_attn(
    const __bf16* __restrict__ Q, const __bf16* __restrict__ K,
    const __bf16* __restrict__ Vt, const int* __restrict__ amask,
    __bf16* __restrict__ O)                               // [b][s][1024]
{
  __shared__ __bf16 plds[8][16][32];                      // per-wave P staging
  const int wave = threadIdx.x >> 5;
  const int lane = threadIdx.x & 31;
  const int bh = blockIdx.y;                              // b*16 + h
  const int b  = bh >> 4;
  const int h  = bh & 15;
  const int i0 = (blockIdx.x * 8 + wave) * 16;
  const int c    = lane & 15;
  const int hb   = lane >> 4;
  const int ak8  = hb * 16;
  const int bk16 = hb * 16;

  const __bf16* Qb = Q  + ((size_t)bh * S_ + i0) * HD;
  const __bf16* Kb = K  + (size_t)bh * S_ * HD;
  const __bf16* Vb = Vt + (size_t)bh * HD * S_;

  // Q tile as two A-fragments (d 0..31, 32..63), resident all loop long
  v16bf qa0 = load_a16(Qb + c * HD + ak8);
  v16bf qa1 = load_a16(Qb + c * HD + 32 + ak8);

  v8f acc[4] = {};                                        // 16x64 f32 output
  float mrow[8], lrow[8];
#pragma unroll
  for (int t = 0; t < 8; ++t) { mrow[t] = -__builtin_inff(); lrow[t] = 0.f; }

  int jstart = i0 - (WIN - 1);
  if (jstart < 0) jstart = 0;
  jstart &= ~31;
  const int jend = i0 + 15;

  // prologue: K fragments for the first key step (clamped tail addresses)
  v16bf kf0[2], kf1[2];
#pragma unroll
  for (int t = 0; t < 2; ++t) {
    int jr = jstart + t * 16 + c;
    if (jr > S_ - 1) jr = S_ - 1;
    const __bf16* kp = Kb + (size_t)jr * HD + bk16;
    kf0[t] = load_b16(kp);
    kf1[t] = load_b16(kp + 32);
  }

  for (int j0 = jstart; j0 <= jend; j0 += 32) {
    // ---- scores: two 16x16 key sub-tiles, 2 WMMAs each over head_dim
    v8f s[2];
#pragma unroll
    for (int t = 0; t < 2; ++t) {
      v8f cs = {};
      cs = wmma_bf16(qa0, kf0[t], cs);
      cs = wmma_bf16(qa1, kf1[t], cs);
      s[t] = cs;
    }
    // ---- prefetch next step's K fragments across the back-edge
#pragma unroll
    for (int t = 0; t < 2; ++t) {
      int jr = j0 + 32 + t * 16 + c;
      if (jr > S_ - 1) jr = S_ - 1;
      const __bf16* kp = Kb + (size_t)jr * HD + bk16;
      kf0[t] = load_b16(kp);
      kf1[t] = load_b16(kp + 32);
    }
    // ---- V fragments for this step (independent of softmax; scheduler hoists)
    v16bf vf[4];
#pragma unroll
    for (int f = 0; f < 4; ++f)
      vf[f] = load_b16(Vb + (size_t)(f * 16 + c) * S_ + j0 + bk16);

    // ---- causal sliding-window + pad mask; per-row max
    float rmax[8];
#pragma unroll
    for (int rr = 0; rr < 8; ++rr) rmax[rr] = -__builtin_inff();
#pragma unroll
    for (int t = 0; t < 2; ++t) {
      const int j = j0 + t * 16 + c;
      const bool kvalid = (j < S_) && (amask[b * S_ + j] != 0);
#pragma unroll
      for (int rr = 0; rr < 8; ++rr) {
        const int i = i0 + rr + hb * 8;
        const bool ok = kvalid && (j <= i) && (j > i - WIN);
        float sv = ok ? s[t][rr] : -__builtin_inff();
        s[t][rr] = sv;
        rmax[rr] = fmaxf(rmax[rr], sv);
      }
    }
    // row reduce across 16-lane halves (rows 0-7 in lanes 0-15, 8-15 in 16-31)
#pragma unroll
    for (int rr = 0; rr < 8; ++rr) {
      float v = rmax[rr];
      v = fmaxf(v, __shfl_xor(v, 1, 16));
      v = fmaxf(v, __shfl_xor(v, 2, 16));
      v = fmaxf(v, __shfl_xor(v, 4, 16));
      v = fmaxf(v, __shfl_xor(v, 8, 16));
      rmax[rr] = v;
    }
    // ---- online softmax rescale factors
    float alpha[8];
#pragma unroll
    for (int rr = 0; rr < 8; ++rr) {
      float mn = fmaxf(mrow[rr], rmax[rr]);
      alpha[rr] = (mn == -__builtin_inff()) ? 1.f : __expf(mrow[rr] - mn);
      mrow[rr] = mn;
    }
    // ---- p = exp(s - m), row sums, stage P into LDS for A-fragment reload
    float rsum[8];
#pragma unroll
    for (int rr = 0; rr < 8; ++rr) rsum[rr] = 0.f;
#pragma unroll
    for (int t = 0; t < 2; ++t)
#pragma unroll
      for (int rr = 0; rr < 8; ++rr) {
        float sv = s[t][rr];
        float p = (sv < -1e37f) ? 0.f : __expf(sv - mrow[rr]);
        rsum[rr] += p;
        plds[wave][rr + hb * 8][t * 16 + c] = f2bf(p);
      }
#pragma unroll
    for (int rr = 0; rr < 8; ++rr) {
      float v = rsum[rr];
      v += __shfl_xor(v, 1, 16);
      v += __shfl_xor(v, 2, 16);
      v += __shfl_xor(v, 4, 16);
      v += __shfl_xor(v, 8, 16);
      lrow[rr] = lrow[rr] * alpha[rr] + v;
    }
#pragma unroll
    for (int f = 0; f < 4; ++f)
#pragma unroll
      for (int rr = 0; rr < 8; ++rr) acc[f][rr] *= alpha[rr];

    // ---- P·V: A = P (16x32 over keys), B columns from transposed V
    v16bf pa = load_a16(&plds[wave][c][ak8]);
#pragma unroll
    for (int f = 0; f < 4; ++f)
      acc[f] = wmma_bf16(pa, vf[f], acc[f]);
  }

  float rl[8];
#pragma unroll
  for (int rr = 0; rr < 8; ++rr) rl[rr] = (lrow[rr] > 0.f) ? (1.f / lrow[rr]) : 0.f;

  __bf16* Ob = O + ((size_t)b * S_ + i0) * HID + h * HD;
#pragma unroll
  for (int f = 0; f < 4; ++f)
#pragma unroll
    for (int rr = 0; rr < 8; ++rr)
      Ob[(size_t)(rr + hb * 8) * HID + f * 16 + c] = f2bf(acc[f][rr] * rl[rr]);
}

// ---------- kernel 4: output projection GEMM (2x2 blocked, pipelined) ----------
// grid = (256, 4), block = 256. One wave -> 32x32 output block (4 WMMA tiles).
__global__ __launch_bounds__(256) void swa_out_gemm(
    const __bf16* __restrict__ Ob, const __bf16* __restrict__ wo,
    float* __restrict__ out)
{
  const int wave = threadIdx.x >> 5;
  const int lane = threadIdx.x & 31;
  const int m0 = blockIdx.x * 32;
  const int n0 = (blockIdx.y * 8 + wave) * 32;
  const int c  = lane & 15;
  const int hb = lane >> 4;
  const __bf16* Ap0 = Ob + (size_t)(m0 + c) * HID + hb * 16;
  const __bf16* Ap1 = Ap0 + (size_t)16 * HID;
  const __bf16* Bp0 = wo + (size_t)(n0 + c) * HID + hb * 16;
  const __bf16* Bp1 = Bp0 + (size_t)16 * HID;

  v8f c00 = {}, c01 = {}, c10 = {}, c11 = {};

  v16bf a0 = load_a16(Ap0);
  v16bf a1 = load_a16(Ap1);
  v16bf b0 = load_b16(Bp0);
  v16bf b1 = load_b16(Bp1);

#pragma unroll 2
  for (int k = 0; k < HID; k += 32) {
    v16bf a0n = load_a16(Ap0 + k + 32);
    v16bf a1n = load_a16(Ap1 + k + 32);
    v16bf b0n = load_b16(Bp0 + k + 32);
    v16bf b1n = load_b16(Bp1 + k + 32);

    c00 = wmma_bf16(a0, b0, c00);
    c01 = wmma_bf16(a0, b1, c01);
    c10 = wmma_bf16(a1, b0, c10);
    c11 = wmma_bf16(a1, b1, c11);

    a0 = a0n; a1 = a1n; b0 = b0n; b1 = b1n;
  }
#pragma unroll
  for (int rr = 0; rr < 8; ++rr) {
    size_t r0 = (size_t)(m0 + rr + hb * 8) * HID;
    size_t r1 = (size_t)(m0 + 16 + rr + hb * 8) * HID;
    out[r0 + n0 + c]      = c00[rr];
    out[r0 + n0 + 16 + c] = c01[rr];
    out[r1 + n0 + c]      = c10[rr];
    out[r1 + n0 + 16 + c] = c11[rr];
  }
}

// ---------- launcher ----------
extern "C" void kernel_launch(void* const* d_in, const int* in_sizes, int n_in,
                              void* d_out, int out_size, void* d_ws, size_t ws_size,
                              hipStream_t stream) {
  (void)in_sizes; (void)n_in; (void)out_size; (void)ws_size;
  const float* x  = (const float*)d_in[0];
  const int*   am = (const int*)d_in[1];
  const float* Wq = (const float*)d_in[2];
  const float* Wk = (const float*)d_in[3];
  const float* Wv = (const float*)d_in[4];
  const float* Wo = (const float*)d_in[5];
  float* out = (float*)d_out;

  char* ws = (char*)d_ws;
  size_t off = 0;
  auto alloc = [&](size_t elems) -> __bf16* {
    __bf16* p = (__bf16*)(ws + off);
    off += ((elems * 2 + 255) & ~(size_t)255);
    return p;
  };
  const size_t NX = (size_t)B_ * S_ * HID;      // 8,388,608
  const size_t NW = (size_t)HID * HID;          // 1,048,576
  const size_t NQ = (size_t)B_ * NH_ * S_ * HD; // 8,388,608

  __bf16* xb = alloc(NX);
  __bf16* wq = alloc(NW);
  __bf16* wk = alloc(NW);
  __bf16* wv = alloc(NW);
  __bf16* wo = alloc(NW);
  __bf16* Qb = alloc(NQ);
  __bf16* Kb = alloc(NQ);
  __bf16* Vt = alloc(NQ);
  __bf16* Ob = alloc(NX);   // keep last so clamped/prefetch tail reads stay in ws

  int n4;
  n4 = (int)(NX / 4); swa_cvt_bf16<<<(n4 + 255) / 256, 256, 0, stream>>>(x,  xb, n4);
  n4 = (int)(NW / 4); swa_cvt_bf16<<<(n4 + 255) / 256, 256, 0, stream>>>(Wq, wq, n4);
  n4 = (int)(NW / 4); swa_cvt_bf16<<<(n4 + 255) / 256, 256, 0, stream>>>(Wk, wk, n4);
  n4 = (int)(NW / 4); swa_cvt_bf16<<<(n4 + 255) / 256, 256, 0, stream>>>(Wv, wv, n4);
  n4 = (int)(NW / 4); swa_cvt_bf16<<<(n4 + 255) / 256, 256, 0, stream>>>(Wo, wo, n4);

  swa_qkv_gemm<<<dim3((B_ * S_) / 32, 8), 256, 0, stream>>>(xb, wq, wk, wv, Qb, Kb, Vt);
  swa_attn    <<<dim3(S_ / 16 / 8, B_ * NH_), 256, 0, stream>>>(Qb, Kb, Vt, am, Ob);
  swa_out_gemm<<<dim3((B_ * S_) / 32, 4), 256, 0, stream>>>(Ob, wo, out);
}
